// RegionProposalNetwork_1881195676021
// MI455X (gfx1250) — compile-verified, matrix operand order checked by reading, not run on hardware
//
#include <hip/hip_runtime.h>
#include <hip/hip_bf16.h>
#include <math.h>

// ---------------------------------------------------------------------------
// Types for CDNA5 WMMA (wave32): v16bf A/B fragments, v8f f32 accumulator.
// ---------------------------------------------------------------------------
typedef __attribute__((ext_vector_type(16))) __bf16 v16bf;
typedef __attribute__((ext_vector_type(8)))  float  v8f;
typedef __attribute__((__vector_size__(16))) int    v4i_async;  // matches builtin param

#define NEGV (-1e9f)

// d_out layout (floats), concatenated reference outputs:
//   rpn_locs  [8][36864][4]  -> [0,        1179648)
//   rpn_scores[8][36864][2]  -> [1179648,  1769472)
//   rois      [8][300][4]    -> [1769472,  1779072)
//   roi_idx   [8][300]       -> [1779072,  1781472)
//   anchor    [36864][4]     -> [1781472,  1928928)
#define OUT_SC   1179648
#define OUT_ROI  1769472
#define OUT_IDX  1779072
#define OUT_ANC  1781472

// ---- gfx1250 async global->LDS copy (ASYNCcnt-tracked), with fallback -----
#if defined(__has_builtin)
#if __has_builtin(__builtin_amdgcn_global_load_async_to_lds_b128)
#define USE_ASYNC_LDS 1
#endif
#if __has_builtin(__builtin_amdgcn_s_wait_asynccnt)
#define HAVE_WAIT_ASYNC_BUILTIN 1
#endif
#endif
#ifndef USE_ASYNC_LDS
#define USE_ASYNC_LDS 0
#endif
#ifndef HAVE_WAIT_ASYNC_BUILTIN
#define HAVE_WAIT_ASYNC_BUILTIN 0
#endif

static __device__ __forceinline__ void asyncCopy16B(const unsigned short* g,
                                                    unsigned short* l) {
#if USE_ASYNC_LDS
  __builtin_amdgcn_global_load_async_to_lds_b128(
      (__attribute__((address_space(1))) v4i_async*)(g),
      (__attribute__((address_space(3))) v4i_async*)(l), 0, 0);
#else
  *(uint4*)l = *(const uint4*)g;
#endif
}

static __device__ __forceinline__ void waitAsync() {
#if USE_ASYNC_LDS
#if HAVE_WAIT_ASYNC_BUILTIN
  __builtin_amdgcn_s_wait_asynccnt(0);
#else
  asm volatile("s_wait_asynccnt 0" ::: "memory");
#endif
#endif
}

static __device__ __forceinline__ unsigned short f2bf(float f) {
  unsigned u = __float_as_uint(f);
  unsigned r = u + 0x7FFFu + ((u >> 16) & 1u);   // round-to-nearest-even
  return (unsigned short)(r >> 16);
}

// float -> monotonically-increasing unsigned key (for radix select)
static __device__ __forceinline__ unsigned fkey(float f) {
  unsigned u = __float_as_uint(f);
  return (u & 0x80000000u) ? ~u : (u | 0x80000000u);
}

union FragBF { v16bf v; uint4 q[2]; };

// Load one 16x32 bf16 WMMA fragment row-set from an LDS tile stored
// row-major with row stride 32 elements (64 B).
// Lane L holds matrix row (L&15); K groups are [8*hi, 8*hi+8) and +16.
static __device__ __forceinline__ v16bf ldsFrag(const unsigned short* rowBase, int hi) {
  FragBF f;
  f.q[0] = *(const uint4*)(rowBase + 8 * hi);
  f.q[1] = *(const uint4*)(rowBase + 8 * hi + 16);
  return f.v;
}

// ---------------------------------------------------------------------------
// Prep: x [8,512,64,64] f32 (NCHW) -> xpad [8,66,66,512] bf16 (padded NHWC)
// ---------------------------------------------------------------------------
__global__ __launch_bounds__(256) void prep_xpad(const float* __restrict__ x,
                                                 unsigned short* __restrict__ xpad) {
  size_t idx = (size_t)blockIdx.x * 256 + threadIdx.x;   // 8*66*66*512 = 17842176
  int ci = (int)(idx & 511);
  size_t t = idx >> 9;
  int xx = (int)(t % 66); t /= 66;
  int yy = (int)(t % 66);
  int n  = (int)(t / 66);
  float v = 0.f;
  if (yy >= 1 && yy <= 64 && xx >= 1 && xx <= 64)
    v = x[(((size_t)n * 512 + ci) * 64 + (yy - 1)) * 64 + (xx - 1)];
  xpad[idx] = f2bf(v);
}

// W1 [512(co),512(ci),3,3] f32 -> w1t [co][k] bf16, k=(ky*3+kx)*512+ci
__global__ __launch_bounds__(256) void prep_w1t(const float* __restrict__ W1,
                                                unsigned short* __restrict__ w1t) {
  size_t idx = (size_t)blockIdx.x * 256 + threadIdx.x;   // 512*4608 = 2359296
  int k  = (int)(idx & 4607);
  int co = (int)(idx / 4608);
  int kyx = k >> 9;
  int ci  = k & 511;
  int ky  = kyx / 3, kx = kyx - ky * 3;
  w1t[idx] = f2bf(W1[(((size_t)co * 512 + ci) * 3 + ky) * 3 + kx]);
}

// Fuse Wl[36,512] + Ws[18,512] (+10 zero rows) -> wc [64][512] bf16, bc[64]
__global__ __launch_bounds__(256) void prep_wc(const float* __restrict__ Wl,
                                               const float* __restrict__ Ws,
                                               const float* __restrict__ bl,
                                               const float* __restrict__ bs,
                                               unsigned short* __restrict__ wc,
                                               float* __restrict__ bc) {
  int idx = blockIdx.x * 256 + threadIdx.x;              // 64*512 = 32768
  int r = idx >> 9, k = idx & 511;
  float v = 0.f;
  if (r < 36)       v = Wl[(size_t)r * 512 + k];
  else if (r < 54)  v = Ws[(size_t)(r - 36) * 512 + k];
  wc[idx] = f2bf(v);
  if (idx < 64) {
    float b = 0.f;
    if (idx < 36)      b = bl[idx];
    else if (idx < 54) b = bs[idx - 36];
    bc[idx] = b;
  }
}

// ---------------------------------------------------------------------------
// Conv 3x3 as implicit GEMM: M=32768 (n,y,x), N=512 (co), K=4608 (ky,kx,ci).
// Block 256 thr = 8 waves, tile 128x128, K-step 32, bf16 WMMA, ReLU epilogue.
// Double-buffered LDS + async global->LDS: ONE barrier per K-step.
// h output: [m][co] bf16 (NHWC).
// ---------------------------------------------------------------------------
__global__ __launch_bounds__(256) void conv3x3_wmma(
    const unsigned short* __restrict__ xpad,   // [8,66,66,512] bf16
    const unsigned short* __restrict__ w1t,    // [512][4608]   bf16
    const float* __restrict__ bias1,           // [512]
    unsigned short* __restrict__ h)            // [32768][512]  bf16
{
  __shared__ unsigned short As[2][128 * 32];
  __shared__ unsigned short Bs[2][128 * 32];

  const int tid  = threadIdx.x;
  const int lane = tid & 31;
  const int wid  = tid >> 5;
  const int m0 = blockIdx.x * 128;
  const int n0 = blockIdx.y * 128;
  const int wm = (wid & 3) * 32;     // 4 waves along M
  const int wn = (wid >> 2) * 64;    // 2 waves along N
  const int hi  = lane >> 4;
  const int l15 = lane & 15;

  const v8f vzero = {0.f, 0.f, 0.f, 0.f, 0.f, 0.f, 0.f, 0.f};
  v8f acc[2][4];
#pragma unroll
  for (int i = 0; i < 2; ++i)
#pragma unroll
    for (int j = 0; j < 4; ++j) acc[i][j] = vzero;

  // Global->LDS staging: each thread moves 8 bf16 (16 B) per matrix per step.
  const int ar = tid >> 1;           // tile row 0..127
  const int ah = tid & 1;            // 16-element half
  const int m  = m0 + ar;
  const int nimg = m >> 12;
  const int yy = (m >> 6) & 63;
  const int xx = m & 63;
  const size_t browB = (size_t)(n0 + ar) * 4608 + (size_t)ah * 16;

  unsigned short* lA[2] = { &As[0][ar * 32 + ah * 16], &As[1][ar * 32 + ah * 16] };
  unsigned short* lB[2] = { &Bs[0][ar * 32 + ah * 16], &Bs[1][ar * 32 + ah * 16] };

  auto gA = [&](int kc) -> const unsigned short* {
    const int k0  = kc << 5;
    const int kyx = k0 >> 9;
    const int ky  = kyx / 3;
    const int kx  = kyx - ky * 3;
    const int ci0 = k0 & 511;
    return xpad + ((size_t)((nimg * 66 + yy + ky) * 66 + (xx + kx)) << 9) + ci0 + ah * 16;
  };

  // Prologue: stage chunk 0 into buffer 0.
  asyncCopy16B(gA(0), lA[0]);
  asyncCopy16B(w1t + browB, lB[0]);

  for (int kc = 0; kc < 144; ++kc) {
    const int cur = kc & 1;
    const int nxt = cur ^ 1;

    waitAsync();        // this wave's chunk-kc LDS writes are complete
    __syncthreads();    // chunk kc visible; buf[nxt] no longer being read

    if (kc + 1 < 144) { // stage chunk kc+1 into the other buffer
      const unsigned short* ga = gA(kc + 1);
      const unsigned short* gb = w1t + browB + ((size_t)(kc + 1) << 5);
      asyncCopy16B(ga, lA[nxt]);
      asyncCopy16B(gb, lB[nxt]);
      __builtin_prefetch(ga + 32, 0, 1);
      __builtin_prefetch(gb + 32, 0, 1);
    }

    const unsigned short* ab = &As[cur][0];
    const unsigned short* bb = &Bs[cur][0];
    v16bf a0  = ldsFrag(ab + (wm + l15) * 32, hi);
    v16bf a1  = ldsFrag(ab + (wm + 16 + l15) * 32, hi);
    v16bf bf0 = ldsFrag(bb + (wn + l15) * 32, hi);
    v16bf bf1 = ldsFrag(bb + (wn + 16 + l15) * 32, hi);
    v16bf bf2 = ldsFrag(bb + (wn + 32 + l15) * 32, hi);
    v16bf bf3 = ldsFrag(bb + (wn + 48 + l15) * 32, hi);

    acc[0][0] = __builtin_amdgcn_wmma_f32_16x16x32_bf16(false, a0, false, bf0, (short)0, acc[0][0], false, false);
    acc[0][1] = __builtin_amdgcn_wmma_f32_16x16x32_bf16(false, a0, false, bf1, (short)0, acc[0][1], false, false);
    acc[0][2] = __builtin_amdgcn_wmma_f32_16x16x32_bf16(false, a0, false, bf2, (short)0, acc[0][2], false, false);
    acc[0][3] = __builtin_amdgcn_wmma_f32_16x16x32_bf16(false, a0, false, bf3, (short)0, acc[0][3], false, false);
    acc[1][0] = __builtin_amdgcn_wmma_f32_16x16x32_bf16(false, a1, false, bf0, (short)0, acc[1][0], false, false);
    acc[1][1] = __builtin_amdgcn_wmma_f32_16x16x32_bf16(false, a1, false, bf1, (short)0, acc[1][1], false, false);
    acc[1][2] = __builtin_amdgcn_wmma_f32_16x16x32_bf16(false, a1, false, bf2, (short)0, acc[1][2], false, false);
    acc[1][3] = __builtin_amdgcn_wmma_f32_16x16x32_bf16(false, a1, false, bf3, (short)0, acc[1][3], false, false);
  }

  // Epilogue: bias + ReLU -> bf16 NHWC h. D layout: lane holds N=l15,
  // VGPR r holds M = r + 8*hi.
#pragma unroll
  for (int i = 0; i < 2; ++i) {
#pragma unroll
    for (int j = 0; j < 4; ++j) {
      const int mb = m0 + wm + i * 16 + hi * 8;
      const int co = n0 + wn + j * 16 + l15;
      const float b = bias1[co];
#pragma unroll
      for (int r = 0; r < 8; ++r) {
        float v = acc[i][j][r] + b;
        v = v > 0.f ? v : 0.f;
        h[(size_t)(mb + r) * 512 + co] = f2bf(v);
      }
    }
  }
}

// ---------------------------------------------------------------------------
// 1x1 heads as one GEMM: M=32768, K=512, N=64 (36 loc + 18 score + 10 pad).
// Epilogue scatters fp32 straight into the transposed output layout.
// ---------------------------------------------------------------------------
__global__ __launch_bounds__(256) void head_wmma(
    const unsigned short* __restrict__ h,    // [32768][512] bf16
    const unsigned short* __restrict__ wc,   // [64][512]    bf16
    const float* __restrict__ bc,            // [64]
    float* __restrict__ out)
{
  __shared__ unsigned short As[128 * 32];
  __shared__ unsigned short Bs[64 * 32];

  const int tid  = threadIdx.x;
  const int lane = tid & 31;
  const int wid  = tid >> 5;
  const int m0 = blockIdx.x * 128;
  const int wm = (wid & 3) * 32;
  const int wn = (wid >> 2) * 32;
  const int hi  = lane >> 4;
  const int l15 = lane & 15;

  const v8f vzero = {0.f, 0.f, 0.f, 0.f, 0.f, 0.f, 0.f, 0.f};
  v8f acc[2][2];
#pragma unroll
  for (int i = 0; i < 2; ++i)
#pragma unroll
    for (int j = 0; j < 2; ++j) acc[i][j] = vzero;

  const int ar = tid >> 1;
  const int ah = tid & 1;

  for (int kc = 0; kc < 16; ++kc) {
    const int k0 = kc << 5;
    *(uint4*)(As + ar * 32 + ah * 16) =
        *(const uint4*)(h + (size_t)(m0 + ar) * 512 + k0 + ah * 16);
    if (tid < 128)
      *(uint4*)(Bs + ar * 32 + ah * 16) =
          *(const uint4*)(wc + (size_t)ar * 512 + k0 + ah * 16);
    __syncthreads();

    v16bf a0  = ldsFrag(As + (wm + l15) * 32, hi);
    v16bf a1  = ldsFrag(As + (wm + 16 + l15) * 32, hi);
    v16bf bf0 = ldsFrag(Bs + (wn + l15) * 32, hi);
    v16bf bf1 = ldsFrag(Bs + (wn + 16 + l15) * 32, hi);

    acc[0][0] = __builtin_amdgcn_wmma_f32_16x16x32_bf16(false, a0, false, bf0, (short)0, acc[0][0], false, false);
    acc[0][1] = __builtin_amdgcn_wmma_f32_16x16x32_bf16(false, a0, false, bf1, (short)0, acc[0][1], false, false);
    acc[1][0] = __builtin_amdgcn_wmma_f32_16x16x32_bf16(false, a1, false, bf0, (short)0, acc[1][0], false, false);
    acc[1][1] = __builtin_amdgcn_wmma_f32_16x16x32_bf16(false, a1, false, bf1, (short)0, acc[1][1], false, false);
    __syncthreads();
  }

#pragma unroll
  for (int i = 0; i < 2; ++i) {
#pragma unroll
    for (int j = 0; j < 2; ++j) {
      const int mb = m0 + wm + i * 16 + hi * 8;
      const int c  = wn + j * 16 + l15;
      const float b = bc[c];
#pragma unroll
      for (int r = 0; r < 8; ++r) {
        const int mm = mb + r;
        const int n = mm >> 12;        // image
        const int q = mm & 4095;       // y*64+x
        const float v = acc[i][j][r] + b;
        if (c < 36)
          out[(size_t)n * 147456 + (size_t)q * 36 + c] = v;             // locs
        else if (c < 54)
          out[OUT_SC + (size_t)n * 73728 + (size_t)q * 18 + (c - 36)] = v; // scores
      }
    }
  }
}

// ---------------------------------------------------------------------------
// Anchors + loc2bbox + clip + min-size mask + fg softmax.
// ---------------------------------------------------------------------------
__global__ __launch_bounds__(256) void proposal_kernel(
    const float* __restrict__ out,   // reads locs/scores
    float* __restrict__ roi,         // [8][36864][4]
    float* __restrict__ sc,          // [8][36864]
    float* __restrict__ outw,        // = out (writable alias for anchor)
    const int* __restrict__ ih, const int* __restrict__ iw)
{
  const int idx = blockIdx.x * 256 + threadIdx.x;   // 8*36864 = 294912
  const int n = idx / 36864;
  const int p = idx - n * 36864;
  const int a = p % 9;
  const int q = p / 9;
  const int gx = q & 63;
  const int gy = q >> 6;

  const float RR[3] = {0.5f, 1.f, 2.f};
  const float SS[3] = {8.f, 16.f, 32.f};
  const float r = RR[a / 3], s = SS[a % 3];
  const float hb = 16.f * s * sqrtf(r);
  const float wb = 16.f * s * sqrtf(1.f / r);
  const float ax1 = gx * 16.f - 0.5f * wb;
  const float ay1 = gy * 16.f - 0.5f * hb;
  const float ax2 = gx * 16.f + 0.5f * wb;
  const float ay2 = gy * 16.f + 0.5f * hb;

  if (n == 0) {
    outw[OUT_ANC + (size_t)p * 4 + 0] = ax1;
    outw[OUT_ANC + (size_t)p * 4 + 1] = ay1;
    outw[OUT_ANC + (size_t)p * 4 + 2] = ax2;
    outw[OUT_ANC + (size_t)p * 4 + 3] = ay2;
  }

  const float* loc = out + (size_t)n * 147456 + (size_t)p * 4;
  const float w  = ax2 - ax1;
  const float hh = ay2 - ay1;
  const float cx = ax1 + 0.5f * w;
  const float cy = ay1 + 0.5f * hh;
  const float ncx = loc[0] * w + cx;
  const float ncy = loc[1] * hh + cy;
  const float nw  = expf(loc[2]) * w;
  const float nh  = expf(loc[3]) * hh;

  const float W = (float)(*iw);
  const float H = (float)(*ih);
  float x1 = fminf(fmaxf(ncx - 0.5f * nw, 0.f), W);
  float y1 = fminf(fmaxf(ncy - 0.5f * nh, 0.f), H);
  float x2 = fminf(fmaxf(ncx + 0.5f * nw, 0.f), W);
  float y2 = fminf(fmaxf(ncy + 0.5f * nh, 0.f), H);

  const float* sp = out + OUT_SC + (size_t)n * 73728 + (size_t)q * 18 + a * 2;
  const float fg = 1.f / (1.f + expf(sp[0] - sp[1]));   // softmax [...,1]

  const bool ok = (x2 - x1 >= 16.f) && (y2 - y1 >= 16.f);

  roi[(size_t)idx * 4 + 0] = x1;
  roi[(size_t)idx * 4 + 1] = y1;
  roi[(size_t)idx * 4 + 2] = x2;
  roi[(size_t)idx * 4 + 3] = y2;
  sc[idx] = ok ? fg : NEGV;
}

// ---------------------------------------------------------------------------
// Exact 3000th-largest score per image via 4-pass byte radix select.
// ---------------------------------------------------------------------------
__global__ __launch_bounds__(256) void radix_select_kernel(
    const float* __restrict__ sc, unsigned* __restrict__ tau)
{
  __shared__ unsigned hist[256];
  __shared__ unsigned s_prefix, s_mask;
  __shared__ int s_need;
  const int tid = threadIdx.x;
  const int img = blockIdx.x;
  const float* p = sc + (size_t)img * 36864;

  if (tid == 0) { s_prefix = 0u; s_mask = 0u; s_need = 3000; }
  __syncthreads();

  for (int pass = 3; pass >= 0; --pass) {
    hist[tid] = 0u;
    __syncthreads();
    const unsigned mask = s_mask, prefix = s_prefix;
    const int shift = pass * 8;
    for (int i = tid; i < 36864; i += 256) {
      unsigned k = fkey(p[i]);
      if ((k & mask) == prefix) atomicAdd(&hist[(k >> shift) & 255u], 1u);
    }
    __syncthreads();
    if (tid == 0) {
      int need = s_need;
      unsigned cum = 0;
      int b = 255;
      for (; b > 0; --b) {
        unsigned c = hist[b];
        if (cum + c >= (unsigned)need) break;
        cum += c;
      }
      s_prefix = prefix | ((unsigned)b << shift);
      s_mask   = mask | (255u << shift);
      s_need   = need - (int)cum;
    }
    __syncthreads();
  }
  if (tid == 0) tau[img] = s_prefix;
}

__global__ __launch_bounds__(256) void mask_kernel(
    const float* __restrict__ sc, const unsigned* __restrict__ tau,
    float* __restrict__ scm)
{
  const int idx = blockIdx.x * 256 + threadIdx.x;   // 294912
  const int n = idx / 36864;
  const float v = sc[idx];
  scm[idx] = (fkey(v) >= tau[n]) ? v : NEGV;
}

// ---------------------------------------------------------------------------
// Greedy NMS, one block per image: 300 x (argmax-reduce + IoU suppress).
// ---------------------------------------------------------------------------
__global__ __launch_bounds__(512) void nms_kernel(
    const float* __restrict__ roi, float* __restrict__ scm,
    float* __restrict__ out)
{
  __shared__ float svals[512];
  __shared__ int   sidx[512];
  __shared__ float sbox[5];
  __shared__ int   sj, svalid;

  const int tid = threadIdx.x;
  const int img = blockIdx.x;
  const float* boxes = roi + (size_t)img * 36864 * 4;
  float* sc = scm + (size_t)img * 36864;

  for (int it = 0; it < 300; ++it) {
    // argmax (first-max-index tie rule, like jnp.argmax)
    float best = -3.4e38f; int bi = 0;
    for (int i = tid; i < 36864; i += 512) {
      float v = sc[i];
      if (v > best) { best = v; bi = i; }
    }
    svals[tid] = best; sidx[tid] = bi;
    __syncthreads();
    for (int off = 256; off > 0; off >>= 1) {
      if (tid < off) {
        float v2 = svals[tid + off]; int i2 = sidx[tid + off];
        if (v2 > svals[tid] || (v2 == svals[tid] && i2 < sidx[tid])) {
          svals[tid] = v2; sidx[tid] = i2;
        }
      }
      __syncthreads();
    }
    if (tid == 0) {
      const int j = sidx[0];
      const float v = svals[0];
      const int valid = v > (NEGV * 0.5f);
      const float* bb = boxes + (size_t)j * 4;
      const float x1 = bb[0], y1 = bb[1], x2 = bb[2], y2 = bb[3];
      sbox[0] = x1; sbox[1] = y1; sbox[2] = x2; sbox[3] = y2;
      sbox[4] = (x2 - x1) * (y2 - y1);
      sj = j; svalid = valid;
      float* rr = out + OUT_ROI + (size_t)img * 1200 + (size_t)it * 4;
      if (valid) { rr[0] = x1; rr[1] = y1; rr[2] = x2; rr[3] = y2; }
      else       { rr[0] = 0.f; rr[1] = 0.f; rr[2] = 0.f; rr[3] = 0.f; }
      out[OUT_IDX + (size_t)img * 300 + it] = valid ? (float)img : -1.f;
    }
    __syncthreads();
    if (svalid) {
      const int j = sj;
      const float bx1 = sbox[0], by1 = sbox[1], bx2 = sbox[2], by2 = sbox[3];
      const float ba = sbox[4];
      for (int i = tid; i < 36864; i += 512) {
        const float* bb = boxes + (size_t)i * 4;
        float xx1 = fmaxf(bb[0], bx1), yy1 = fmaxf(bb[1], by1);
        float xx2 = fminf(bb[2], bx2), yy2 = fminf(bb[3], by2);
        float inter = fmaxf(xx2 - xx1, 0.f) * fmaxf(yy2 - yy1, 0.f);
        float ai = (bb[2] - bb[0]) * (bb[3] - bb[1]);
        float iou = inter / (ai + ba - inter + 1e-9f);
        if (iou > 0.7f || i == j) sc[i] = NEGV;
      }
    }
    __syncthreads();
  }
}

// ---------------------------------------------------------------------------
// Launcher. Workspace layout (bytes, all 256-aligned):
//   xpad bf16 35,684,352 | w1t bf16 4,718,592 | wc 65,536 | bc 256 |
//   h bf16 33,554,432 | roi f32 4,718,592 | sc 1,179,648 | scm 1,179,648 |
//   tau 32   => ~81.1 MB total.
// ---------------------------------------------------------------------------
extern "C" void kernel_launch(void* const* d_in, const int* in_sizes, int n_in,
                              void* d_out, int out_size, void* d_ws, size_t ws_size,
                              hipStream_t stream) {
  (void)in_sizes; (void)n_in; (void)out_size; (void)ws_size;
  const float* x  = (const float*)d_in[0];
  const float* W1 = (const float*)d_in[1];
  const float* b1 = (const float*)d_in[2];
  const float* Ws = (const float*)d_in[3];
  const float* bs = (const float*)d_in[4];
  const float* Wl = (const float*)d_in[5];
  const float* bl = (const float*)d_in[6];
  const int* ih   = (const int*)d_in[7];
  const int* iw   = (const int*)d_in[8];
  float* out = (float*)d_out;

  char* ws = (char*)d_ws;
  unsigned short* xpad = (unsigned short*)(ws + 0);
  unsigned short* w1t  = (unsigned short*)(ws + 35684352);
  unsigned short* wc   = (unsigned short*)(ws + 40402944);
  float*          bc   = (float*)(ws + 40468480);
  unsigned short* h    = (unsigned short*)(ws + 40468736);
  float*          roi  = (float*)(ws + 74023168);
  float*          sc   = (float*)(ws + 78741760);
  float*          scm  = (float*)(ws + 79921408);
  unsigned*       tau  = (unsigned*)(ws + 81101056);

  prep_xpad<<<69696, 256, 0, stream>>>(x, xpad);
  prep_w1t<<<9216, 256, 0, stream>>>(W1, w1t);
  prep_wc<<<128, 256, 0, stream>>>(Wl, Ws, bl, bs, wc, bc);
  conv3x3_wmma<<<dim3(256, 4), 256, 0, stream>>>(xpad, w1t, b1, h);
  head_wmma<<<256, 256, 0, stream>>>(h, wc, bc, out);
  proposal_kernel<<<1152, 256, 0, stream>>>(out, roi, sc, out, ih, iw);
  radix_select_kernel<<<8, 256, 0, stream>>>(sc, tau);
  mask_kernel<<<1152, 256, 0, stream>>>(sc, tau, scm);
  nms_kernel<<<8, 512, 0, stream>>>(roi, scm, out);
}
